// HGAT_65317862638060
// MI455X (gfx1250) — matrix operand order repaired and checked by reading, not compiled, not run on hardware
//
#include <hip/hip_runtime.h>
#include <hip/hip_bf16.h>
#include <math.h>

// ---------- constants from the reference ----------
#define HEADS 3
#define DDIM  64
#define HD    192      // HEADS*DDIM
#define INF_  256
#define EFD   64
#define NETT  8
#define NEGS  0.2f
#define EPS_  1e-12f

// ---------- vector types for WMMA ----------
typedef __attribute__((ext_vector_type(16))) __bf16       v16bf;
typedef __attribute__((ext_vector_type(8)))  __bf16       v8bf;
typedef __attribute__((ext_vector_type(8)))  float        v8f;
typedef __attribute__((ext_vector_type(8)))  unsigned short v8us;

// ---------- small helpers ----------
__device__ __forceinline__ unsigned short f2bf(float f) {
    unsigned u = __float_as_uint(f);
    unsigned r = (u + 0x7FFFu + ((u >> 16) & 1u)) >> 16;   // RNE
    return (unsigned short)r;
}

__device__ __forceinline__ void atomicMaxF(float* addr, float v) {
    // sign-split trick -> native global_atomic_max_i32 / global_atomic_min_u32
    if (v >= 0.0f) atomicMax((int*)addr, __float_as_int(v));
    else           atomicMin((unsigned int*)addr, (unsigned int)__float_as_int(v));
}

__device__ __forceinline__ float eluf(float x) {
    return x > 0.0f ? x : (__expf(x) - 1.0f);
}

// =====================================================================
// fill kernel (workspace init each call; harness does not re-poison)
// =====================================================================
__global__ void fill_u32_kernel(unsigned* __restrict__ p, unsigned val, int n) {
    int t = blockIdx.x * blockDim.x + threadIdx.x;
    if (t < n) p[t] = val;
}

// =====================================================================
// float -> bf16 conversion
// =====================================================================
__global__ void cvt_bf16_kernel(const float* __restrict__ x,
                                unsigned short* __restrict__ y, int n) {
    int t = blockIdx.x * blockDim.x + threadIdx.x;
    if (t < n) y[t] = f2bf(x[t]);
}

// =====================================================================
// W[K,Nc] (f32, row-major) -> Wt[Nc,K] (bf16): WMMA B fragments become
// two contiguous 16B loads per lane.
// =====================================================================
__global__ void wt_bf16_kernel(const float* __restrict__ W,
                               unsigned short* __restrict__ Wt, int K, int Nc) {
    int t = blockIdx.x * blockDim.x + threadIdx.x;
    if (t >= K * Nc) return;
    int n = t / K, k = t % K;
    Wt[t] = f2bf(W[(size_t)k * Nc + n]);
}

// =====================================================================
// ee table: ee[t,h] = sum_f (Eemb[t]@We)[h*EF+f] * ae[h,f]   (24 values)
// =====================================================================
__global__ void ee_tab_kernel(const float* __restrict__ Eemb,
                              const float* __restrict__ We,
                              const float* __restrict__ ae,
                              float* __restrict__ tab) {
    int t = threadIdx.x;
    if (t >= NETT * HEADS) return;
    int ty = t / HEADS, h = t % HEADS;
    float s = 0.0f;
    for (int f = 0; f < EFD; ++f) {
        float v = 0.0f;
        for (int k = 0; k < EFD; ++k)
            v += Eemb[ty * EFD + k] * We[k * (HEADS * EFD) + h * EFD + f];
        s += v * ae[h * EFD + f];
    }
    tab[t] = s;
}

// =====================================================================
// WMMA GEMM: C[M,Nc] = A[M,K](bf16) @ B[K,Nc], B given as Bt[Nc,K](bf16).
// One wave per 16x16 tile of C, 8 waves per block.
// Fragment layouts follow CDNA5 ISA 7.12.2 (16-bit A 16x32, B 32x16).
// =====================================================================
__global__ __launch_bounds__(256) void gemm_bf16_wmma_kernel(
    const unsigned short* __restrict__ A,
    const unsigned short* __restrict__ Bt,
    float* __restrict__ C, int M, int K, int Nc)
{
    const int wave  = threadIdx.x >> 5;
    const int lane  = threadIdx.x & 31;
    const int mtile = blockIdx.x * 8 + wave;
    const int ntile = blockIdx.y;
    const int Mtiles = (M + 15) >> 4;
    if (mtile >= Mtiles) return;                 // wave-uniform exit

    const int half = lane >> 4;
    const int l16  = lane & 15;

    int arow = mtile * 16 + l16; if (arow >= M) arow = M - 1;
    const int bcol = ntile * 16 + l16;

    const unsigned short* arowp = A  + (size_t)arow * K;
    const unsigned short* bcolp = Bt + (size_t)bcol * K;

    v8f acc = {};
    for (int kb = 0; kb < K; kb += 32) {
        // A: elements 0..7 -> K = kb+8*half .. +7 ; 8..15 -> K = kb+16+8*half ..
        v8bf a0 = __builtin_bit_cast(v8bf, *(const v8us*)(arowp + kb + 8 * half));
        v8bf a1 = __builtin_bit_cast(v8bf, *(const v8us*)(arowp + kb + 16 + 8 * half));
        // B: same K order as A: elements 0..7 -> K = kb+8*half .. +7 ; 8..15 -> K = kb+16+8*half ..
        v8bf b0 = __builtin_bit_cast(v8bf, *(const v8us*)(bcolp + kb + 8 * half));
        v8bf b1 = __builtin_bit_cast(v8bf, *(const v8us*)(bcolp + kb + 16 + 8 * half));
        v16bf av = __builtin_shufflevector(a0, a1, 0,1,2,3,4,5,6,7,8,9,10,11,12,13,14,15);
        v16bf bv = __builtin_shufflevector(b0, b1, 0,1,2,3,4,5,6,7,8,9,10,11,12,13,14,15);
        acc = __builtin_amdgcn_wmma_f32_16x16x32_bf16(
                  false, av, false, bv, (short)0, acc, false, false);
    }

    // C/D layout: acc[r] -> row = mtile*16 + half*8 + r, col = ntile*16 + l16
    const int row0 = mtile * 16 + half * 8;
    float* cp = C + (size_t)row0 * Nc + ntile * 16 + l16;
    #pragma unroll
    for (int r = 0; r < 8; ++r) {
        if (row0 + r < M) cp[(size_t)r * Nc] = acc[r];
    }
}

// =====================================================================
// el[n,h] = <feat[n,h,:], al[h,:]>, er likewise
// =====================================================================
__global__ void elr_kernel(const float* __restrict__ feat,
                           const float* __restrict__ al,
                           const float* __restrict__ ar,
                           float* __restrict__ el, float* __restrict__ er, int n3) {
    int t = blockIdx.x * blockDim.x + threadIdx.x;
    if (t >= n3) return;
    int n = t / HEADS, h = t % HEADS;
    const float* f   = feat + (size_t)n * HD + h * DDIM;
    const float* alh = al + h * DDIM;
    const float* arh = ar + h * DDIM;
    float sl = 0.0f, sr = 0.0f;
    #pragma unroll 8
    for (int d = 0; d < DDIM; ++d) { sl += f[d] * alh[d]; sr += f[d] * arh[d]; }
    el[t] = sl; er[t] = sr;
}

// =====================================================================
// edge pass 1: logit = leakyrelu(el[src]+er[dst]+ee); atomic segment max
// =====================================================================
__global__ void edge_logit_kernel(const int* __restrict__ src,
                                  const int* __restrict__ dst,
                                  const int* __restrict__ ef,
                                  const float* __restrict__ el,
                                  const float* __restrict__ er,
                                  const float* __restrict__ eetab,
                                  float* __restrict__ logit,
                                  float* __restrict__ mx, int E3) {
    int t = blockIdx.x * blockDim.x + threadIdx.x;
    if (t >= E3) return;
    int e = t / HEADS, h = t % HEADS;
    int s = src[e], d = dst[e];
    float x = el[s * HEADS + h] + er[d * HEADS + h] + eetab[ef[e] * HEADS + h];
    x = (x > 0.0f) ? x : NEGS * x;
    logit[t] = x;
    atomicMaxF(mx + d * HEADS + h, x);
}

// =====================================================================
// edge pass 2: a = exp(logit - mx[dst]); atomic segment sum into den
// =====================================================================
__global__ void edge_expsum_kernel(const int* __restrict__ dst,
                                   const float* __restrict__ mx,
                                   float* __restrict__ logit,
                                   float* __restrict__ den, int E3) {
    int t = blockIdx.x * blockDim.x + threadIdx.x;
    if (t >= E3) return;
    int e = t / HEADS, h = t % HEADS;
    int d = dst[e];
    float a = __expf(logit[t] - mx[d * HEADS + h]);
    logit[t] = a;
    unsafeAtomicAdd(den + d * HEADS + h, a);
}

// =====================================================================
// edge pass 3: rst[dst,h,:] += (a/den) * feat[src,h,:]
// One wave per edge; 192 components / 32 lanes = 6 per lane.
// feat (38 MB) and rst live in the 192 MB L2 -> L2-resident atomics.
// =====================================================================
__global__ __launch_bounds__(256) void aggregate_kernel(
    const int* __restrict__ src, const int* __restrict__ dst,
    const float* __restrict__ a, const float* __restrict__ den,
    const float* __restrict__ feat, float* __restrict__ rst, int E)
{
    int wave = threadIdx.x >> 5, lane = threadIdx.x & 31;
    int e = blockIdx.x * 8 + wave;
    if (e >= E) return;
    int s = src[e], d = dst[e];
    const float* fs = feat + (size_t)s * HD;
    float*       rd = rst  + (size_t)d * HD;
    __builtin_prefetch(fs, 0, 1);
    float w0 = a[e * 3 + 0] / fmaxf(den[d * 3 + 0], EPS_);
    float w1 = a[e * 3 + 1] / fmaxf(den[d * 3 + 1], EPS_);
    float w2 = a[e * 3 + 2] / fmaxf(den[d * 3 + 2], EPS_);
    #pragma unroll
    for (int i = 0; i < 6; ++i) {
        int c = lane + i * 32;
        float w = (c < 64) ? w0 : ((c < 128) ? w1 : w2);
        unsafeAtomicAdd(rd + c, w * fs[c]);
    }
}

// =====================================================================
// layer-1 epilogue: h1bf = bf16(elu(rst))
// =====================================================================
__global__ void elu_bf16_kernel(const float* __restrict__ rst,
                                unsigned short* __restrict__ h1bf, int n) {
    int t = blockIdx.x * blockDim.x + threadIdx.x;
    if (t < n) h1bf[t] = f2bf(eluf(rst[t]));
}

// =====================================================================
// final: out[n,:] = normalize(mean_h elu(rst2 + res2))
// =====================================================================
__global__ __launch_bounds__(64) void finalize_kernel(
    const float* __restrict__ rst, const float* __restrict__ res,
    float* __restrict__ out)
{
    int n = blockIdx.x, d = threadIdx.x;
    float v = 0.0f;
    #pragma unroll
    for (int h = 0; h < HEADS; ++h) {
        size_t idx = (size_t)n * HD + h * DDIM + d;
        v += eluf(rst[idx] + res[idx]);
    }
    v *= (1.0f / (float)HEADS);
    __shared__ float red[64];
    red[d] = v * v;
    __syncthreads();
    #pragma unroll
    for (int s = 32; s > 0; s >>= 1) {
        if (d < s) red[d] += red[d + s];
        __syncthreads();
    }
    float nrm = sqrtf(red[0]);
    out[(size_t)n * DDIM + d] = v / fmaxf(nrm, EPS_);
}

// =====================================================================
// host-side launcher
// =====================================================================
extern "C" void kernel_launch(void* const* d_in, const int* in_sizes, int n_in,
                              void* d_out, int out_size, void* d_ws, size_t ws_size,
                              hipStream_t stream) {
    const float* x     = (const float*)d_in[0];
    const int*   src   = (const int*)d_in[1];
    const int*   dst   = (const int*)d_in[2];
    const int*   efeat = (const int*)d_in[3];
    const float* W1    = (const float*)d_in[4];
    const float* Eemb1 = (const float*)d_in[5];
    const float* We1   = (const float*)d_in[6];
    const float* al1   = (const float*)d_in[7];
    const float* ar1   = (const float*)d_in[8];
    const float* ae1   = (const float*)d_in[9];
    const float* W2    = (const float*)d_in[10];
    const float* Eemb2 = (const float*)d_in[11];
    const float* We2   = (const float*)d_in[12];
    const float* al2   = (const float*)d_in[13];
    const float* ar2   = (const float*)d_in[14];
    const float* ae2   = (const float*)d_in[15];
    const float* Wres2 = (const float*)d_in[16];
    float* out = (float*)d_out;

    const int N = in_sizes[0] / INF_;
    const int E = in_sizes[1];

    // ---- workspace carve-out ----
    char* ws = (char*)d_ws;
    size_t off = 0;
    auto alloc = [&](size_t bytes) -> void* {
        off = (off + 255) & ~(size_t)255;
        void* p = ws + off;
        off += bytes;
        return p;
    };
    unsigned short* Xbf     = (unsigned short*)alloc((size_t)N * INF_ * 2);
    unsigned short* H1bf    = (unsigned short*)alloc((size_t)N * HD * 2);
    unsigned short* Wt1     = (unsigned short*)alloc((size_t)HD * INF_ * 2);
    unsigned short* Wt2     = (unsigned short*)alloc((size_t)HD * HD * 2);
    unsigned short* Wtr2    = (unsigned short*)alloc((size_t)HD * HD * 2);
    float* feat  = (float*)alloc((size_t)N * HD * 4);   // reused for layer 2
    float* res2  = (float*)alloc((size_t)N * HD * 4);
    float* rst   = (float*)alloc((size_t)N * HD * 4);   // reused per layer
    float* logit = (float*)alloc((size_t)E * HEADS * 4);
    float* el    = (float*)alloc((size_t)N * HEADS * 4);
    float* er    = (float*)alloc((size_t)N * HEADS * 4);
    float* mx    = (float*)alloc((size_t)N * HEADS * 4);
    float* den   = (float*)alloc((size_t)N * HEADS * 4);
    float* tab1  = (float*)alloc(NETT * HEADS * 4);
    float* tab2  = (float*)alloc(NETT * HEADS * 4);

    const int TB = 256;
    const int N3 = N * HEADS;
    const int E3 = E * HEADS;
    const int Mtiles = (N + 15) / 16;
    dim3 ggrid((Mtiles + 7) / 8, HD / 16);

    auto cdiv = [](int a, int b) { return (a + b - 1) / b; };

    // ---- setup: ee tables, weight transposes, x -> bf16 ----
    ee_tab_kernel<<<1, 32, 0, stream>>>(Eemb1, We1, ae1, tab1);
    ee_tab_kernel<<<1, 32, 0, stream>>>(Eemb2, We2, ae2, tab2);
    wt_bf16_kernel<<<cdiv(HD * INF_, TB), TB, 0, stream>>>(W1, Wt1, INF_, HD);
    wt_bf16_kernel<<<cdiv(HD * HD, TB), TB, 0, stream>>>(W2, Wt2, HD, HD);
    wt_bf16_kernel<<<cdiv(HD * HD, TB), TB, 0, stream>>>(Wres2, Wtr2, HD, HD);
    cvt_bf16_kernel<<<cdiv(N * INF_, TB), TB, 0, stream>>>(x, Xbf, N * INF_);

    // ================= layer 1 =================
    fill_u32_kernel<<<cdiv(N3, TB), TB, 0, stream>>>((unsigned*)mx, 0xFF800000u, N3);
    fill_u32_kernel<<<cdiv(N3, TB), TB, 0, stream>>>((unsigned*)den, 0u, N3);
    fill_u32_kernel<<<cdiv(N * HD, TB), TB, 0, stream>>>((unsigned*)rst, 0u, N * HD);

    gemm_bf16_wmma_kernel<<<ggrid, TB, 0, stream>>>(Xbf, Wt1, feat, N, INF_, HD);
    elr_kernel<<<cdiv(N3, TB), TB, 0, stream>>>(feat, al1, ar1, el, er, N3);
    edge_logit_kernel<<<cdiv(E3, TB), TB, 0, stream>>>(src, dst, efeat, el, er, tab1,
                                                       logit, mx, E3);
    edge_expsum_kernel<<<cdiv(E3, TB), TB, 0, stream>>>(dst, mx, logit, den, E3);
    aggregate_kernel<<<cdiv(E, 8), TB, 0, stream>>>(src, dst, logit, den, feat, rst, E);
    elu_bf16_kernel<<<cdiv(N * HD, TB), TB, 0, stream>>>(rst, H1bf, N * HD);

    // ================= layer 2 =================
    fill_u32_kernel<<<cdiv(N3, TB), TB, 0, stream>>>((unsigned*)mx, 0xFF800000u, N3);
    fill_u32_kernel<<<cdiv(N3, TB), TB, 0, stream>>>((unsigned*)den, 0u, N3);
    fill_u32_kernel<<<cdiv(N * HD, TB), TB, 0, stream>>>((unsigned*)rst, 0u, N * HD);

    gemm_bf16_wmma_kernel<<<ggrid, TB, 0, stream>>>(H1bf, Wt2, feat, N, HD, HD);
    gemm_bf16_wmma_kernel<<<ggrid, TB, 0, stream>>>(H1bf, Wtr2, res2, N, HD, HD);
    elr_kernel<<<cdiv(N3, TB), TB, 0, stream>>>(feat, al2, ar2, el, er, N3);
    edge_logit_kernel<<<cdiv(E3, TB), TB, 0, stream>>>(src, dst, efeat, el, er, tab2,
                                                       logit, mx, E3);
    edge_expsum_kernel<<<cdiv(E3, TB), TB, 0, stream>>>(dst, mx, logit, den, E3);
    aggregate_kernel<<<cdiv(E, 8), TB, 0, stream>>>(src, dst, logit, den, feat, rst, E);

    // ---- final: residual + elu + head-mean + L2 normalize ----
    finalize_kernel<<<N, 64, 0, stream>>>(rst, res2, out);
}
